// DCRNNModel_50483045597455
// MI455X (gfx1250) — matrix operand order repaired and checked by reading, not compiled, not run on hardware
//
#include <hip/hip_runtime.h>
#include <hip/hip_bf16.h>
#include <cstdint>

#define NN 50000
#define EE 800000
#define FF 64
#define HID 128
#define AS 384          // Abuf stride in bf16 elems ([x|fwd|bwd], max 3*128)

typedef __attribute__((ext_vector_type(16))) __bf16 v16bf;
typedef __attribute__((ext_vector_type(8)))  float  v8f;
typedef __attribute__((ext_vector_type(4)))  unsigned int v4u;
typedef __attribute__((ext_vector_type(8)))  int v8i;
typedef __attribute__((ext_vector_type(4)))  int v4i;

#if __has_builtin(__builtin_amdgcn_tensor_load_to_lds) && __has_builtin(__builtin_amdgcn_s_wait_tensorcnt)
#define USE_TDM 1
#else
#define USE_TDM 0
#endif

union ABfrag { v16bf v; uint4 q[2]; };

__device__ __forceinline__ unsigned short f2bf(float f) {
    unsigned u = __builtin_bit_cast(unsigned, f);
    u += 0x7FFFu + ((u >> 16) & 1u);   // round-to-nearest-even
    return (unsigned short)(u >> 16);
}

__device__ __forceinline__ void atomAddF(float* p, float v) {
    __hip_atomic_fetch_add(p, v, __ATOMIC_RELAXED, __HIP_MEMORY_SCOPE_AGENT);
}

// ---------------- utility ----------------
__global__ void zero_kernel(float* p, int count) {
    for (int i = blockIdx.x * blockDim.x + threadIdx.x; i < count; i += gridDim.x * blockDim.x)
        p[i] = 0.0f;
}

// ---------------- degrees + edge norm ----------------
__global__ void degree_kernel(const int* __restrict__ src, const int* __restrict__ dst,
                              const float* __restrict__ ew, float* deg_out, float* deg_in) {
    int e = blockIdx.x * blockDim.x + threadIdx.x;
    if (e >= EE) return;
    float w = ew[e];
    atomAddF(&deg_out[src[e]], w);
    atomAddF(&deg_in[dst[e]], w);
}

__global__ void normw_kernel(const int* __restrict__ src, const int* __restrict__ dst,
                             const float* __restrict__ ew,
                             const float* __restrict__ deg_out, const float* __restrict__ deg_in,
                             float* __restrict__ nwo, float* __restrict__ nwi) {
    int e = blockIdx.x * blockDim.x + threadIdx.x;
    if (e >= EE) return;
    float w = ew[e];
    float dO = deg_out[src[e]];
    float dI = deg_in[dst[e]];
    nwo[e] = dO > 0.0f ? w / dO : 0.0f;
    nwi[e] = dI > 0.0f ? w / dI : 0.0f;
}

// ---------------- weight packing into WMMA B-fragment layout ----------------
// W shape (2,2,cfull,128). Wcat rows: [0,C): W[0,0]+W[1,0]; [C,2C): W[0,1]; [2C,3C): W[1,1].
// Fragment (kt,nt) blob: 32 lanes x 16 bf16; lane l, elem j -> K = kt*32 + ((l&16)>>1) + (j&7) + ((j&8)<<1), N = nt*16 + (l&15).
__global__ void pack_kernel(const float* __restrict__ W, int C, int cfull,
                            unsigned short* __restrict__ out, int total) {
    int idx = blockIdx.x * blockDim.x + threadIdx.x;
    if (idx >= total) return;
    int j    = idx & 15;
    int lane = (idx >> 4) & 31;
    int frag = idx >> 9;
    int kt = frag >> 3, nt = frag & 7;
    int K  = kt * 32 + ((lane & 16) >> 1) + (j & 7) + ((j & 8) << 1);
    int nn = nt * 16 + (lane & 15);
    float v;
    if (K < C)            v = W[(size_t)(0 * cfull + K) * 128 + nn] + W[(size_t)(2 * cfull + K) * 128 + nn];
    else if (K < 2 * C)   v = W[(size_t)(1 * cfull + (K - C)) * 128 + nn];
    else                  v = W[(size_t)(3 * cfull + (K - 2 * C)) * 128 + nn];
    out[idx] = f2bf(v);
}

// ---------------- bf16 staging ----------------
__global__ void stage_x_kernel(const float* __restrict__ x, unsigned short* __restrict__ Abuf,
                               int C, int total) {
    int i = blockIdx.x * blockDim.x + threadIdx.x;
    if (i >= total) return;
    int n = i / C, c = i - n * C;
    Abuf[(size_t)n * AS + c] = f2bf(x[(size_t)n * C + c]);
}

__global__ void cvt_acc_kernel(const float* __restrict__ fwd, const float* __restrict__ bwd,
                               unsigned short* __restrict__ Abuf, int C, int total) {
    int i = blockIdx.x * blockDim.x + threadIdx.x;
    if (i >= total) return;
    int n = i / C, c = i - n * C;
    Abuf[(size_t)n * AS + C + c]     = f2bf(fwd[(size_t)n * C + c]);
    Abuf[(size_t)n * AS + 2 * C + c] = f2bf(bwd[(size_t)n * C + c]);
}

// ---------------- sparse diffusion scatter ----------------
// tpe = C/4 threads per edge; each thread handles 4 columns for both directions.
__global__ void scatter_kernel(const float* __restrict__ xin,
                               const int* __restrict__ src, const int* __restrict__ dst,
                               const float* __restrict__ nwo, const float* __restrict__ nwi,
                               float* __restrict__ fwd, float* __restrict__ bwd,
                               int C, int tpe, int total) {
    int tid = blockIdx.x * blockDim.x + threadIdx.x;
    if (tid >= total) return;
    int e = tid / tpe;
    int t = (tid - e * tpe) * 4;
    int s = src[e], d = dst[e];
    float wo = nwo[e], wi = nwi[e];

    float4 a = *(const float4*)(xin + (size_t)s * C + t);
    float* fp = fwd + (size_t)d * C + t;
    atomAddF(fp + 0, wo * a.x); atomAddF(fp + 1, wo * a.y);
    atomAddF(fp + 2, wo * a.z); atomAddF(fp + 3, wo * a.w);

    float4 b = *(const float4*)(xin + (size_t)d * C + t);
    float* bp = bwd + (size_t)s * C + t;
    atomAddF(bp + 0, wi * b.x); atomAddF(bp + 1, wi * b.y);
    atomAddF(bp + 2, wi * b.z); atomAddF(bp + 3, wi * b.w);
}

// ---------------- fused dual-gate WMMA GEMM + GRU-style epilogue ----------------
// out[n,:] = (1 - sigmoid(A@Wz + bz)) * tanh(A@Wh + bh), A = [x|fwd|bwd] bf16 (N x Kcat)
// A tile (16 rows x Kcat bf16) is staged to LDS by the Tensor Data Mover (TDM) when available.
__global__ __launch_bounds__(256) void gate_kernel(const unsigned short* __restrict__ Abuf,
                                                   const unsigned short* __restrict__ Wzp,
                                                   const unsigned short* __restrict__ Whp,
                                                   const float* __restrict__ bz,
                                                   const float* __restrict__ bh,
                                                   float* __restrict__ out, int Kcat) {
    __shared__ __align__(16) unsigned short lds[16 * AS];   // rows stored with stride == Kcat
    const int m0  = blockIdx.x * 16;
    const int tid = threadIdx.x;

#if USE_TDM
    if (tid < 32) {   // wave 0 issues one TDM DMA for the whole block
        unsigned long long ga = (unsigned long long)(uintptr_t)(Abuf + (size_t)m0 * AS);
        unsigned lds_addr = (unsigned)(uintptr_t)&lds[0];    // low 32 bits = LDS byte offset
        // D# group0: count=1 | lds_addr | global_addr[56:0] | type=2
        v4u g0 = { 1u,
                   lds_addr,
                   (unsigned)(ga & 0xFFFFFFFFu),
                   (unsigned)((ga >> 32) & 0x01FFFFFFu) | (2u << 30) };
        // D# group1: data_size=2B; tensor_dim0=AS; tensor_dim1=NN; tile_dim0=Kcat;
        //            tile_dim1=16; tile_dim2=0; tensor_dim0_stride=AS
        v8i g1 = { (int)(1u << 16),
                   (int)((AS & 0xFFFFu) << 16),
                   (int)(((unsigned)AS >> 16) | ((NN & 0xFFFFu) << 16)),
                   (int)(((unsigned)NN >> 16) | ((unsigned)Kcat << 16)),
                   16,
                   AS,
                   0, 0 };
        v4i z4 = { 0, 0, 0, 0 };
        v8i z8 = { 0, 0, 0, 0, 0, 0, 0, 0 };
        __builtin_amdgcn_tensor_load_to_lds(g0, g1, z4, z4, z8, 0);
        __builtin_amdgcn_s_wait_tensorcnt(0);
    }
    __syncthreads();
#else
    const int kchunks = Kcat >> 3;   // 16-byte chunks per row
    for (int i = tid; i < 16 * kchunks; i += 256) {
        int r = i / kchunks, kc = i - r * kchunks;
        *(uint4*)&lds[r * Kcat + kc * 8] = *(const uint4*)&Abuf[(size_t)(m0 + r) * AS + kc * 8];
    }
    __syncthreads();
#endif

    const int lane = tid & 31;
    const int nt   = tid >> 5;              // wave id -> 16-col tile
    v8f cz = {}; v8f ch = {};
    const int row  = lane & 15;
    const int koff = (lane & 16) >> 1;      // lanes 16-31 start at K+8
    const unsigned short* lrow = &lds[row * Kcat + koff];

    const int ksteps = Kcat >> 5;
    for (int kt = 0; kt < ksteps; ++kt) {
        ABfrag a, fz, fh;
        a.q[0] = *(const uint4*)&lrow[kt * 32];
        a.q[1] = *(const uint4*)&lrow[kt * 32 + 16];
        int fb = (((kt * 8 + nt) * 32) + lane) * 16;
        fz.q[0] = *(const uint4*)&Wzp[fb]; fz.q[1] = *(const uint4*)&Wzp[fb + 8];
        fh.q[0] = *(const uint4*)&Whp[fb]; fh.q[1] = *(const uint4*)&Whp[fb + 8];
        cz = __builtin_amdgcn_wmma_f32_16x16x32_bf16(false, a.v, false, fz.v, (short)0, cz, false, false);
        ch = __builtin_amdgcn_wmma_f32_16x16x32_bf16(false, a.v, false, fh.v, (short)0, ch, false, false);
    }

    const int ncol = nt * 16 + (lane & 15);
    const float bzv = bz[ncol];
    const float bhv = bh[ncol];
    const int mbase = m0 + ((lane >> 4) << 3);
#pragma unroll
    for (int g = 0; g < 8; ++g) {
        float z  = 1.0f / (1.0f + __expf(-(cz[g] + bzv)));
        float ht = tanhf(ch[g] + bhv);
        out[(size_t)(mbase + g) * 128 + ncol] = (1.0f - z) * ht;
    }
}

// ---------------- final linear head: one wave per node ----------------
__global__ void head_kernel(const float* __restrict__ h, const float* __restrict__ Wlin,
                            const float* __restrict__ blin, float* __restrict__ out) {
    int wid  = (blockIdx.x * blockDim.x + threadIdx.x) >> 5;
    int lane = threadIdx.x & 31;
    if (wid >= NN) return;
    float4 hv = *(const float4*)(h + (size_t)wid * 128 + lane * 4);
    float4 wv = *(const float4*)(Wlin + lane * 4);
    float p = hv.x * wv.x + hv.y * wv.y + hv.z * wv.z + hv.w * wv.w;
#pragma unroll
    for (int off = 16; off > 0; off >>= 1) p += __shfl_xor(p, off, 32);
    if (lane == 0) out[wid] = p + blin[0];
}

// =======================================================================
extern "C" void kernel_launch(void* const* d_in, const int* in_sizes, int n_in,
                              void* d_out, int out_size, void* d_ws, size_t ws_size,
                              hipStream_t stream) {
    const float* x    = (const float*)d_in[0];
    const int*   ei   = (const int*)d_in[1];
    const float* ew   = (const float*)d_in[2];
    const float* W1z  = (const float*)d_in[3];  const float* b1z = (const float*)d_in[4];
    const float* W1h  = (const float*)d_in[7];  const float* b1h = (const float*)d_in[8];
    const float* W2z  = (const float*)d_in[9];  const float* b2z = (const float*)d_in[10];
    const float* W2h  = (const float*)d_in[13]; const float* b2h = (const float*)d_in[14];
    const float* Wlin = (const float*)d_in[15]; const float* blin = (const float*)d_in[16];
    float* out = (float*)d_out;
    const int* src = ei;
    const int* dst = ei + EE;

    // workspace carve (256B aligned)
    char* w = (char*)d_ws;
    auto alloc = [&](size_t bytes) -> char* {
        char* p = w; w += (bytes + 255) & ~(size_t)255; return p;
    };
    float* deg    = (float*)alloc((size_t)2 * NN * 4);             // deg_out | deg_in
    float* deg_o  = deg;
    float* deg_i  = deg + NN;
    float* nwo    = (float*)alloc((size_t)EE * 4);
    float* nwi    = (float*)alloc((size_t)EE * 4);
    float* acc    = (float*)alloc((size_t)2 * NN * 128 * 4);       // fwd | bwd (stride C)
    float* hbuf   = (float*)alloc((size_t)NN * 128 * 4);
    unsigned short* Abuf = (unsigned short*)alloc((size_t)NN * AS * 2);
    unsigned short* Wzp1 = (unsigned short*)alloc((size_t)192 * 128 * 2);
    unsigned short* Whp1 = (unsigned short*)alloc((size_t)192 * 128 * 2);
    unsigned short* Wzp2 = (unsigned short*)alloc((size_t)384 * 128 * 2);
    unsigned short* Whp2 = (unsigned short*)alloc((size_t)384 * 128 * 2);

    const int TB = 256;
    auto gridOf = [](long long n, int b) { return (int)((n + b - 1) / b); };

    // 1) degrees + edge normalization
    zero_kernel<<<gridOf(2 * NN, TB), TB, 0, stream>>>(deg, 2 * NN);
    degree_kernel<<<gridOf(EE, TB), TB, 0, stream>>>(src, dst, ew, deg_o, deg_i);
    normw_kernel<<<gridOf(EE, TB), TB, 0, stream>>>(src, dst, ew, deg_o, deg_i, nwo, nwi);

    // 2) pack weights (bf16 WMMA fragment layout)
    pack_kernel<<<gridOf(192 * 128, TB), TB, 0, stream>>>(W1z, FF, 192, Wzp1, 192 * 128);
    pack_kernel<<<gridOf(192 * 128, TB), TB, 0, stream>>>(W1h, FF, 192, Whp1, 192 * 128);
    pack_kernel<<<gridOf(384 * 128, TB), TB, 0, stream>>>(W2z, HID, 256, Wzp2, 384 * 128);
    pack_kernel<<<gridOf(384 * 128, TB), TB, 0, stream>>>(W2h, HID, 256, Whp2, 384 * 128);

    // ---- Cell 1: input x (C = 64) ----
    {
        const int C = FF, tpe = C / 4;
        float* fwd = acc;
        float* bwd = acc + (size_t)NN * C;
        zero_kernel<<<gridOf((long long)2 * NN * C, TB), TB, 0, stream>>>(acc, 2 * NN * C);
        stage_x_kernel<<<gridOf((long long)NN * C, TB), TB, 0, stream>>>(x, Abuf, C, NN * C);
        scatter_kernel<<<gridOf((long long)EE * tpe, TB), TB, 0, stream>>>(
            x, src, dst, nwo, nwi, fwd, bwd, C, tpe, EE * tpe);
        cvt_acc_kernel<<<gridOf((long long)NN * C, TB), TB, 0, stream>>>(fwd, bwd, Abuf, C, NN * C);
        gate_kernel<<<NN / 16, 256, 0, stream>>>(Abuf, Wzp1, Whp1, b1z, b1h, hbuf, 3 * C);
    }

    // ---- Cell 2: input hbuf (C = 128) ----
    {
        const int C = HID, tpe = C / 4;
        float* fwd = acc;
        float* bwd = acc + (size_t)NN * C;
        zero_kernel<<<gridOf((long long)2 * NN * C, TB), TB, 0, stream>>>(acc, 2 * NN * C);
        stage_x_kernel<<<gridOf((long long)NN * C, TB), TB, 0, stream>>>(hbuf, Abuf, C, NN * C);
        scatter_kernel<<<gridOf((long long)EE * tpe, TB), TB, 0, stream>>>(
            hbuf, src, dst, nwo, nwi, fwd, bwd, C, tpe, EE * tpe);
        cvt_acc_kernel<<<gridOf((long long)NN * C, TB), TB, 0, stream>>>(fwd, bwd, Abuf, C, NN * C);
        gate_kernel<<<NN / 16, 256, 0, stream>>>(Abuf, Wzp2, Whp2, b2z, b2h, hbuf, 3 * C);
    }

    // 3) linear head
    head_kernel<<<gridOf((long long)NN * 32, TB), TB, 0, stream>>>(hbuf, Wlin, blin, out);
}
